// Decoder_Attention_46411416600755
// MI455X (gfx1250) — matrix operand order repaired and checked by reading, not compiled, not run on hardware
//
#include <hip/hip_runtime.h>
#include <hip/hip_bf16.h>

// ---------------- problem constants ----------------
#define BB 128
#define SS 64
#define HH 256
#define EE 256
#define VV 16000
#define TT 32
#define START_TOK 1
#define G4H 1024        // 4*H
#define XK 512          // E + H (combined A-matrix width for gates GEMM)

typedef __bf16 v16bf __attribute__((ext_vector_type(16)));
typedef float  v8f   __attribute__((ext_vector_type(8)));

// ---------------- helpers ----------------
__device__ __forceinline__ unsigned short f32_to_bf16(float f) {
  union { float f; unsigned int u; } v; v.f = f;
  unsigned int u = v.u;
  u += 0x7FFFu + ((u >> 16) & 1u);   // round-to-nearest-even
  return (unsigned short)(u >> 16);
}

__device__ __forceinline__ float sigmoidf_(float x) {
  return 1.0f / (1.0f + expf(-x));
}

__device__ __forceinline__ v16bf make_frag(uint4 q0, uint4 q1) {
  union { v16bf v; uint4 q[2]; } u;
  u.q[0] = q0; u.q[1] = q1;
  return u.v;
}

// ---------------- one-time kernels ----------------

__global__ void k_f32_to_bf16(const float* __restrict__ src,
                              unsigned short* __restrict__ dst, long long n) {
  long long i = (long long)blockIdx.x * blockDim.x + threadIdx.x;
  long long stride = (long long)gridDim.x * blockDim.x;
  for (; i < n; i += stride) dst[i] = f32_to_bf16(src[i]);
}

// Wg[g, 0:256] = W_ih[g, 0:256]; Wg[g, 256:512] = W_hh[g, 0:256]   (bf16)
__global__ void k_build_wg(const float* __restrict__ W_ih,
                           const float* __restrict__ W_hh,
                           unsigned short* __restrict__ Wg) {
  long long i = (long long)blockIdx.x * blockDim.x + threadIdx.x;
  long long stride = (long long)gridDim.x * blockDim.x;
  const long long n = (long long)G4H * XK;
  for (; i < n; i += stride) {
    int g = (int)(i >> 9);
    int k = (int)(i & 511);
    float v = (k < EE) ? W_ih[(size_t)g * XK + k] : W_hh[(size_t)g * HH + (k - EE)];
    Wg[i] = f32_to_bf16(v);
  }
}

// u[h] = sum_g V_attn[g] * U_attn[g, h]
__global__ void k_u(const float* __restrict__ U_attn,
                    const float* __restrict__ V_attn,
                    float* __restrict__ u) {
  int h = threadIdx.x;   // 256 threads, 1 block
  float acc = 0.f;
  for (int g = 0; g < HH; ++g) acc += V_attn[g] * U_attn[(size_t)g * HH + h];
  u[h] = acc;
}

// Per-batch: scores = enc . u, softmax over S, context, and write attentions
// (weights are step-invariant: tanh(hW) term is constant in s -> drops in softmax)
__global__ void k_attn(const float* __restrict__ enc,
                       const float* __restrict__ u,
                       float* __restrict__ context,
                       float* __restrict__ attn_out) {
  const int b = blockIdx.x;
  const int tid = threadIdx.x;
  __shared__ float sw[SS];
  __shared__ float sinv;

  if (tid < SS) {
    const float* row = enc + ((size_t)b * SS + tid) * HH;
    float acc = 0.f;
    for (int h = 0; h < HH; ++h) acc += row[h] * u[h];
    sw[tid] = acc;
  }
  __syncthreads();
  if (tid == 0) {
    float mx = sw[0];
    for (int s = 1; s < SS; ++s) mx = fmaxf(mx, sw[s]);
    float sum = 0.f;
    for (int s = 0; s < SS; ++s) { float e = expf(sw[s] - mx); sw[s] = e; sum += e; }
    sinv = 1.0f / sum;
  }
  __syncthreads();
  const float inv = sinv;

  {
    const int h = tid;
    float acc = 0.f;
    for (int s = 0; s < SS; ++s) acc += sw[s] * enc[((size_t)b * SS + s) * HH + h];
    context[(size_t)b * HH + h] = acc * inv;
  }
  for (int idx = tid; idx < TT * SS; idx += blockDim.x)
    attn_out[(size_t)b * TT * SS + idx] = sw[idx & (SS - 1)] * inv;
}

// ctx_gates[b, g] = sum_h context[b,h] * W_ih[g, E + h]    (f32, one-time)
__global__ void k_ctx_gates(const float* __restrict__ context,
                            const float* __restrict__ W_ih,
                            float* __restrict__ ctxg) {
  int idx = blockIdx.x * blockDim.x + threadIdx.x;   // 128*1024
  int b = idx >> 10;
  int g = idx & 1023;
  float acc = 0.f;
  const float* wrow = W_ih + (size_t)g * XK + EE;
  const float* crow = context + (size_t)b * HH;
  for (int h = 0; h < HH; ++h) acc += crow[h] * wrow[h];
  ctxg[idx] = acc;
}

// init: c = enc_c, h_f32 = enc_h, X[:,E:] = bf16(enc_h)
__global__ void k_init(const float* __restrict__ enc_h,
                       const float* __restrict__ enc_c,
                       float* __restrict__ c_f32,
                       float* __restrict__ h_f32,
                       unsigned short* __restrict__ X) {
  int idx = blockIdx.x * blockDim.x + threadIdx.x;   // B*H
  int b = idx >> 8;
  int hh = idx & 255;
  c_f32[idx] = enc_c[idx];
  float hv = enc_h[idx];
  h_f32[idx] = hv;
  X[(size_t)b * XK + EE + hh] = f32_to_bf16(hv);
}

// ---------------- per-step kernels ----------------

// X[b, 0:E] = bf16(relu(embedding[tok_t[b]]))
__global__ void k_prep(const int* __restrict__ target,
                       const float* __restrict__ embedding,
                       unsigned short* __restrict__ X, int t) {
  const int b = blockIdx.x;
  const int e = threadIdx.x;
  int tok = (t == 0) ? START_TOK : target[(size_t)b * TT + (t - 1)];
  float v = embedding[(size_t)tok * EE + e];
  X[(size_t)b * XK + e] = f32_to_bf16(fmaxf(v, 0.0f));
}

// ---------------------------------------------------------------
// Double-buffered WMMA GEMM core.
//   A: [128, pitchA] bf16, rows = batch; wave strip = 32 rows (2 M-tiles)
//   B: [N, pitchB]  bf16 N-major (row n holds K contiguous); 4 N-tiles/wave
//   Per-lane base pointers + compile-time immediate offsets only.
// ---------------------------------------------------------------
#define GEMM_CORE(Abase, pitchA, Bbase, pitchB, NKITER)                        \
  v8f acc[2][4];                                                               \
  _Pragma("unroll") for (int i = 0; i < 2; ++i)                                \
  _Pragma("unroll") for (int j = 0; j < 4; ++j)                                \
  _Pragma("unroll") for (int r = 0; r < 8; ++r) acc[i][j][r] = 0.0f;           \
  uint4 aq[2][2][2];                                                           \
  uint4 bq[2][4][2];                                                           \
  /* prologue: load k-iter 0 into buffer 0 */                                  \
  _Pragma("unroll") for (int i = 0; i < 2; ++i) {                              \
    aq[0][i][0] = *reinterpret_cast<const uint4*>(Abase + i * 16 * (pitchA));  \
    aq[0][i][1] = *reinterpret_cast<const uint4*>(Abase + i * 16 * (pitchA) + 16); \
  }                                                                            \
  _Pragma("unroll") for (int j = 0; j < 4; ++j) {                              \
    bq[0][j][0] = *reinterpret_cast<const uint4*>(Bbase + j * 16 * (pitchB));  \
    bq[0][j][1] = *reinterpret_cast<const uint4*>(Bbase + j * 16 * (pitchB) + 16); \
  }                                                                            \
  _Pragma("unroll") for (int kk = 0; kk < (NKITER); ++kk) {                    \
    const int buf = kk & 1;                                                    \
    const int nbuf = buf ^ 1;                                                  \
    if (kk + 1 < (NKITER)) {                                                   \
      const int k1 = (kk + 1) * 32;                                            \
      _Pragma("unroll") for (int i = 0; i < 2; ++i) {                          \
        aq[nbuf][i][0] = *reinterpret_cast<const uint4*>(Abase + i * 16 * (pitchA) + k1); \
        aq[nbuf][i][1] = *reinterpret_cast<const uint4*>(Abase + i * 16 * (pitchA) + k1 + 16); \
      }                                                                        \
      _Pragma("unroll") for (int j = 0; j < 4; ++j) {                          \
        bq[nbuf][j][0] = *reinterpret_cast<const uint4*>(Bbase + j * 16 * (pitchB) + k1); \
        bq[nbuf][j][1] = *reinterpret_cast<const uint4*>(Bbase + j * 16 * (pitchB) + k1 + 16); \
      }                                                                        \
    }                                                                          \
    v16bf a0 = make_frag(aq[buf][0][0], aq[buf][0][1]);                        \
    v16bf a1 = make_frag(aq[buf][1][0], aq[buf][1][1]);                        \
    _Pragma("unroll") for (int j = 0; j < 4; ++j) {                            \
      v16bf bfj = make_frag(bq[buf][j][0], bq[buf][j][1]);                     \
      acc[0][j] = __builtin_amdgcn_wmma_f32_16x16x32_bf16(false, a0, false, bfj, \
                                                          (short)0, acc[0][j], false, false); \
      acc[1][j] = __builtin_amdgcn_wmma_f32_16x16x32_bf16(false, a1, false, bfj, \
                                                          (short)0, acc[1][j], false, false); \
    }                                                                          \
  }

// gates = X @ Wg^T + ctxg + b_ih + b_hh     M=128 N=1024 K=512, WMMA bf16
__global__ void __launch_bounds__(256, 1)
k_gates_gemm(const unsigned short* __restrict__ X,
             const unsigned short* __restrict__ Wg,
             const float* __restrict__ ctxg,
             const float* __restrict__ b_ih,
             const float* __restrict__ b_hh,
             float* __restrict__ gates) {
  const int wave = threadIdx.x >> 5;
  const int lane = threadIdx.x & 31;
  const int wm = wave >> 1;          // 0..3 -> 32-row strip
  const int wn = wave & 1;           // 0..1 -> 64-col strip
  const int nblock = blockIdx.x * 128;
  const int half = lane >> 4;
  const int nl = lane & 15;

  // per-lane base pointers (all fragment addresses are constant offsets)
  const unsigned short* aBase = X + (size_t)(wm * 32 + nl) * XK + half * 8;
  const unsigned short* bBase = Wg + (size_t)(nblock + wn * 64 + nl) * XK + half * 8;

  GEMM_CORE(aBase, XK, bBase, XK, XK / 32)

#pragma unroll
  for (int i = 0; i < 2; ++i) {
#pragma unroll
    for (int j = 0; j < 4; ++j) {
      int g = nblock + wn * 64 + j * 16 + nl;
      float bias = b_ih[g] + b_hh[g];
      int mbase = wm * 32 + i * 16 + 8 * half;
#pragma unroll
      for (int r = 0; r < 8; ++r) {
        int b = mbase + r;
        gates[(size_t)b * G4H + g] = acc[i][j][r] + ctxg[(size_t)b * G4H + g] + bias;
      }
    }
  }
}

// LSTM pointwise: c2 = sig(f)*c + sig(i)*tanh(g); h2 = sig(o)*tanh(c2)
__global__ void k_lstm(const float* __restrict__ gates,
                       float* __restrict__ c_f32,
                       float* __restrict__ h_f32,
                       unsigned short* __restrict__ X) {
  const int b = blockIdx.x;
  const int hh = threadIdx.x;
  const size_t gb = (size_t)b * G4H;
  float gi = gates[gb + hh];
  float gf = gates[gb + HH + hh];
  float gg = gates[gb + 2 * HH + hh];
  float go = gates[gb + 3 * HH + hh];
  const int idx = b * HH + hh;
  float c = c_f32[idx];
  float c2 = sigmoidf_(gf) * c + sigmoidf_(gi) * tanhf(gg);
  float h2 = sigmoidf_(go) * tanhf(c2);
  c_f32[idx] = c2;
  h_f32[idx] = h2;
  X[(size_t)b * XK + EE + hh] = f32_to_bf16(h2);
}

// out[:, t, :] = h2 @ W_out^T + b_out    M=128 N=16000 K=256, WMMA bf16
__global__ void __launch_bounds__(256, 1)
k_out_gemm(const unsigned short* __restrict__ X,   // A = X[:, E:]
           const unsigned short* __restrict__ Wout, // [V][H] bf16
           const float* __restrict__ b_out,
           float* __restrict__ dec_out, int t) {
  const int wave = threadIdx.x >> 5;
  const int lane = threadIdx.x & 31;
  const int wm = wave >> 1;
  const int wn = wave & 1;
  const int nblock = blockIdx.x * 128;    // 125 blocks * 128 = 16000
  const int half = lane >> 4;
  const int nl = lane & 15;

  const unsigned short* aBase = X + EE + (size_t)(wm * 32 + nl) * XK + half * 8;
  const unsigned short* bBase = Wout + (size_t)(nblock + wn * 64 + nl) * HH + half * 8;

  GEMM_CORE(aBase, XK, bBase, HH, HH / 32)

#pragma unroll
  for (int i = 0; i < 2; ++i) {
#pragma unroll
    for (int j = 0; j < 4; ++j) {
      int v = nblock + wn * 64 + j * 16 + nl;
      float bias = b_out[v];
      int mbase = wm * 32 + i * 16 + 8 * half;
#pragma unroll
      for (int r = 0; r < 8; ++r) {
        int b = mbase + r;
        dec_out[((size_t)b * TT + t) * VV + v] = acc[i][j][r] + bias;
      }
    }
  }
}

// final hidden state -> d_out segment
__global__ void k_final_h(const float* __restrict__ h_f32, float* __restrict__ out) {
  int idx = blockIdx.x * blockDim.x + threadIdx.x;   // B*H
  out[idx] = h_f32[idx];
}

// ---------------- host launcher ----------------
extern "C" void kernel_launch(void* const* d_in, const int* in_sizes, int n_in,
                              void* d_out, int out_size, void* d_ws, size_t ws_size,
                              hipStream_t stream) {
  const float* enc      = (const float*)d_in[0];   // [B,S,H]
  const float* enc_h    = (const float*)d_in[1];   // [1,B,H]
  const float* enc_c    = (const float*)d_in[2];   // [1,B,H]
  const int*   target   = (const int*)  d_in[3];   // [B,T]
  const float* embed    = (const float*)d_in[4];   // [V,E]
  // d_in[5] = W_attn : mathematically unused (softmax shift-invariance)
  const float* U_attn   = (const float*)d_in[6];   // [H,H]
  const float* V_attn   = (const float*)d_in[7];   // [1,H]
  const float* W_ih     = (const float*)d_in[8];   // [4H, E+H]
  const float* W_hh     = (const float*)d_in[9];   // [4H, H]
  const float* b_ih     = (const float*)d_in[10];  // [4H]
  const float* b_hh     = (const float*)d_in[11];  // [4H]
  const float* W_out    = (const float*)d_in[12];  // [V, H]
  const float* b_out    = (const float*)d_in[13];  // [V]

  float* dec_out  = (float*)d_out;                          // [B,T,V]
  float* h_out    = dec_out + (size_t)BB * TT * VV;         // [1,B,H]
  float* attn_out = h_out + (size_t)BB * HH;                // [B,T,S]

  char* ws = (char*)d_ws;
  size_t off = 0;
  unsigned short* wout_bf = (unsigned short*)(ws + off); off += (size_t)VV * HH * 2;   // 8.19 MB
  unsigned short* wg_bf   = (unsigned short*)(ws + off); off += (size_t)G4H * XK * 2;  // 1.05 MB
  unsigned short* X       = (unsigned short*)(ws + off); off += (size_t)BB * XK * 2;   // 128 KB
  float* c_f32    = (float*)(ws + off); off += (size_t)BB * HH * 4;
  float* h_f32    = (float*)(ws + off); off += (size_t)BB * HH * 4;
  float* gates    = (float*)(ws + off); off += (size_t)BB * G4H * 4;
  float* ctxg     = (float*)(ws + off); off += (size_t)BB * G4H * 4;
  float* context  = (float*)(ws + off); off += (size_t)BB * HH * 4;
  float* u_vec    = (float*)(ws + off); off += 1024;

  // ---- one-time setup ----
  k_f32_to_bf16<<<2048, 256, 0, stream>>>(W_out, wout_bf, (long long)VV * HH);
  k_build_wg<<<1024, 256, 0, stream>>>(W_ih, W_hh, wg_bf);
  k_u<<<1, 256, 0, stream>>>(U_attn, V_attn, u_vec);
  k_attn<<<BB, 256, 0, stream>>>(enc, u_vec, context, attn_out);
  k_ctx_gates<<<(BB * G4H) / 256, 256, 0, stream>>>(context, W_ih, ctxg);
  k_init<<<BB, 256, 0, stream>>>(enc_h, enc_c, c_f32, h_f32, X);

  // ---- recurrence: T sequential steps ----
  for (int t = 0; t < TT; ++t) {
    k_prep<<<BB, 256, 0, stream>>>(target, embed, X, t);
    k_gates_gemm<<<G4H / 128, 256, 0, stream>>>(X, wg_bf, ctxg, b_ih, b_hh, gates);
    k_lstm<<<BB, 256, 0, stream>>>(gates, c_f32, h_f32, X);
    k_out_gemm<<<VV / 128, 256, 0, stream>>>(X, wout_bf, b_out, dec_out, t);
  }

  k_final_h<<<BB, 256, 0, stream>>>(h_f32, h_out);
}
